// ConcatModel_28003186770374
// MI455X (gfx1250) — compile-verified, matrix operand order checked by reading, not run on hardware
//
#include <hip/hip_runtime.h>
#include <hip/hip_bf16.h>
#include <stdint.h>

typedef __attribute__((ext_vector_type(16))) _Float16 v16h;
typedef __attribute__((ext_vector_type(8)))  _Float16 v8h;
typedef __attribute__((ext_vector_type(8)))  float    v8f;
typedef __attribute__((ext_vector_type(4)))  int      v4i;

#define HDIM   512
#define GDIM   2048   // 4H
#define TSTEPS 64
#define BATCH  256
#define NCOMB  512    // 2B
#define NSTATE 1024   // fw + bw states processed together
#define EDIM   300
#define EPAD   320    // E padded to multiple of 32 for WMMA K
#define DOUT   3
#define EPS    1e-5f

// GEMM tiling: block = 256 thr / 8 waves; block tile 64M x 256N; wave tile 32M x 64N
#define BM   64
#define BN   256
#define BK   32
#define LDSS 40       // LDS row stride in halves: 80B -> conflict-free b128 ds loads

// ---------------------------------------------------------------------------
// gfx1250 async global->LDS path (ASYNCcnt).  The builtin expects
// (int4 __device__* src, int4 __shared__* dst, imm offset, imm cpol)
// per the clang diagnostic from the previous round.
// ---------------------------------------------------------------------------
#if defined(__HIP_DEVICE_COMPILE__) && \
    __has_builtin(__builtin_amdgcn_global_load_async_to_lds_b128) && \
    __has_builtin(__builtin_amdgcn_s_wait_asynccnt)
#define HAVE_ASYNC_LDS 1
typedef __attribute__((address_space(1))) v4i* as1_v4i;
typedef __attribute__((address_space(3))) v4i* as3_v4i;
__device__ __forceinline__ void async_cp16(void* lds, const void* g) {
    __builtin_amdgcn_global_load_async_to_lds_b128(
        (as1_v4i)(void*)g, (as3_v4i)lds, 0, 0);
}
#else
#define HAVE_ASYNC_LDS 0
#endif

__device__ __forceinline__ v8f wmma16(v16h a, v16h b, v8f c) {
    return __builtin_amdgcn_wmma_f32_16x16x32_f16(false, a, false, b, (short)0, c,
                                                  false, false);
}

// Fragment load from an LDS tile stored [row][LDSS halves]:
// lane l<16 -> row row0+l, K = [0..7],[16..23]; lane l>=16 -> K = [8..15],[24..31]
__device__ __forceinline__ v16h frag_lds(const _Float16* s, int row0) {
    int lane = threadIdx.x & 31;
    int l = lane & 15, hi = lane >> 4;
    const _Float16* p = s + (row0 + l) * LDSS + hi * 8;
    v8h lo = *(const v8h*)p;
    v8h hh = *(const v8h*)(p + 16);
    v16h r;
#pragma unroll
    for (int i = 0; i < 8; ++i) { r[i] = lo[i]; r[i + 8] = hh[i]; }
    return r;
}

// ---------------------------------------------------------------------------
// C[M,N] = A[M,K] * W[N,K]^T with double-buffered LDS staging.
// MODE 0: store f16 (x_proj)
// MODE 1: LSTM step epilogue: + x_proj[t | T-1-t] + b_ih + b_hh, store f32
// MODE 2: + bias, ReLU, store f32 (MLP layers)
// ---------------------------------------------------------------------------
template <int MODE>
__global__ __launch_bounds__(256)
void wmma_gemm(const _Float16* __restrict__ A, int lda,
               const _Float16* __restrict__ W, int ldw,
               void* __restrict__ Cout, int ldc, int K,
               const float* __restrict__ bias0,
               const float* __restrict__ bias1,
               const _Float16* __restrict__ xproj, int t) {
    __shared__ __align__(16) _Float16 As[2][BM * LDSS];   // 10 KB
    __shared__ __align__(16) _Float16 Bs[2][BN * LDSS];   // 40 KB

    const int tid  = threadIdx.x;
    const int wave = tid >> 5;
    const int wrow = (wave >> 2) * 32;   // wave M offset within block tile
    const int wcol = (wave & 3) * 64;    // wave N offset within block tile
    const int row0 = blockIdx.x * BM;
    const int col0 = blockIdx.y * BN;

    // staging assignment: 256 threads; A tile = 64 rows x 4 16B-chunks,
    // B tile = 256 cols x 4 16B-chunks (4 per thread)
    const int sRow = tid >> 2;           // 0..63
    const int sChk = (tid & 3) * 8;      // half offset 0,8,16,24

    v8f acc[2][4];
#pragma unroll
    for (int i = 0; i < 2; ++i)
#pragma unroll
        for (int j = 0; j < 4; ++j) acc[i][j] = (v8f){};

    const int nk = K / BK;
    int buf = 0;

#if HAVE_ASYNC_LDS
    auto stage = [&](int k0, int b) {
        async_cp16(&As[b][sRow * LDSS + sChk],
                   A + (size_t)(row0 + sRow) * lda + k0 + sChk);
#pragma unroll
        for (int i = 0; i < 4; ++i) {
            int c = sRow + i * 64;
            async_cp16(&Bs[b][c * LDSS + sChk],
                       W + (size_t)(col0 + c) * ldw + k0 + sChk);
        }
    };
    stage(0, 0);
    for (int kc = 0; kc < nk; ++kc) {
        __builtin_amdgcn_s_wait_asynccnt(0);   // this wave's DMA into buf done
        __syncthreads();                       // everyone's DMA visible
        if (kc + 1 < nk) stage((kc + 1) * BK, buf ^ 1);  // overlap next DMA
        v16h a0 = frag_lds(As[buf], wrow);
        v16h a1 = frag_lds(As[buf], wrow + 16);
#pragma unroll
        for (int j = 0; j < 4; ++j) {
            v16h bf = frag_lds(Bs[buf], wcol + j * 16);
            acc[0][j] = wmma16(a0, bf, acc[0][j]);
            acc[1][j] = wmma16(a1, bf, acc[1][j]);
        }
        buf ^= 1;
    }
#else
    v8h ra, rb[4];
    auto ldg = [&](int k0) {
        ra = *(const v8h*)(A + (size_t)(row0 + sRow) * lda + k0 + sChk);
#pragma unroll
        for (int i = 0; i < 4; ++i)
            rb[i] = *(const v8h*)(W + (size_t)(col0 + sRow + i * 64) * ldw + k0 + sChk);
    };
    auto stg = [&](int b) {
        *(v8h*)&As[b][sRow * LDSS + sChk] = ra;
#pragma unroll
        for (int i = 0; i < 4; ++i)
            *(v8h*)&Bs[b][(sRow + i * 64) * LDSS + sChk] = rb[i];
    };
    ldg(0);
    stg(0);
    for (int kc = 0; kc < nk; ++kc) {
        __syncthreads();                       // stores to buf visible
        if (kc + 1 < nk) ldg((kc + 1) * BK);   // global->reg overlaps compute
        v16h a0 = frag_lds(As[buf], wrow);
        v16h a1 = frag_lds(As[buf], wrow + 16);
#pragma unroll
        for (int j = 0; j < 4; ++j) {
            v16h bf = frag_lds(Bs[buf], wcol + j * 16);
            acc[0][j] = wmma16(a0, bf, acc[0][j]);
            acc[1][j] = wmma16(a1, bf, acc[1][j]);
        }
        __syncthreads();                       // all reads of buf^1 done
        if (kc + 1 < nk) stg(buf ^ 1);
        buf ^= 1;
    }
#endif

    const int lane = tid & 31;
    const int l = lane & 15, hi = lane >> 4;
#pragma unroll
    for (int i = 0; i < 2; ++i) {
#pragma unroll
        for (int j = 0; j < 4; ++j) {
            int col = col0 + wcol + j * 16 + l;
#pragma unroll
            for (int r = 0; r < 8; ++r) {
                int row = row0 + wrow + i * 16 + hi * 8 + r;
                float v = acc[i][j][r];
                if (MODE == 0) {
                    ((_Float16*)Cout)[(size_t)row * ldc + col] = (_Float16)v;
                } else if (MODE == 1) {
                    int dir = row >> 9;            // 0 = forward, 1 = backward
                    int n   = row & (NCOMB - 1);
                    int tt  = dir ? (TSTEPS - 1 - t) : t;
                    v += bias0[col] + bias1[col];
                    v += (float)xproj[((size_t)tt * NCOMB + n) * GDIM + col];
                    ((float*)Cout)[(size_t)row * ldc + col] = v;
                } else {
                    v += bias0[col];
                    v = v > 0.f ? v : 0.f;
                    ((float*)Cout)[(size_t)row * ldc + col] = v;
                }
            }
        }
    }
}

// ---------------------------------------------------------------------------
// fp32 -> f16 weight convert with K-padding (zeros past K)
// ---------------------------------------------------------------------------
__global__ void cvt_f16_pad(const float* __restrict__ src, _Float16* __restrict__ dst,
                            int rows, int K, int Kp) {
    size_t idx = (size_t)blockIdx.x * blockDim.x + threadIdx.x;
    size_t total = (size_t)rows * Kp;
    if (idx >= total) return;
    int k = (int)(idx % Kp);
    size_t r = idx / Kp;
    dst[idx] = (_Float16)((k < K) ? src[r * K + k] : 0.f);
}

// ---------------------------------------------------------------------------
// Embedding gather -> padded f16 X[T*2B, EPAD]
// ---------------------------------------------------------------------------
__global__ void gather_embed(const int* __restrict__ prem, const int* __restrict__ hyp,
                             const float* __restrict__ embed, _Float16* __restrict__ X) {
    size_t idx = (size_t)blockIdx.x * blockDim.x + threadIdx.x;
    size_t total = (size_t)TSTEPS * NCOMB * EPAD;
    if (idx >= total) return;
    int e = (int)(idx % EPAD);
    size_t row = idx / EPAD;
    int n = (int)(row % NCOMB);
    int t = (int)(row / NCOMB);
    float v = 0.f;
    if (e < EDIM) {
        int tok = (n < BATCH) ? prem[t * BATCH + n] : hyp[t * BATCH + (n - BATCH)];
        v = embed[(size_t)tok * EDIM + e];
    }
    X[idx] = (_Float16)v;
}

// ---------------------------------------------------------------------------
// LSTM pointwise cell update (gate order i,f,g,o along 4H)
// ---------------------------------------------------------------------------
__global__ void lstm_pointwise(const float* __restrict__ G, float* __restrict__ c,
                               _Float16* __restrict__ hOut) {
    int idx = blockIdx.x * blockDim.x + threadIdx.x;   // NSTATE*HDIM
    int n = idx >> 9;
    int h = idx & (HDIM - 1);
    const float* g = G + (size_t)n * GDIM;
    float gi = g[h], gf = g[HDIM + h], gg = g[2 * HDIM + h], go = g[3 * HDIM + h];
    float si = 1.f / (1.f + __expf(-gi));
    float sf = 1.f / (1.f + __expf(-gf));
    float so = 1.f / (1.f + __expf(-go));
    float cc = sf * c[idx] + si * tanhf(gg);
    c[idx] = cc;
    hOut[idx] = (_Float16)(so * tanhf(cc));
}

// ---------------------------------------------------------------------------
// feat[B,4H] = [h_fw[:B], h_bw[:B], h_fw[B:], h_bw[B:]]
// ---------------------------------------------------------------------------
__global__ void build_feat(const _Float16* __restrict__ hS, float* __restrict__ feat) {
    int idx = blockIdx.x * blockDim.x + threadIdx.x;   // BATCH*GDIM
    int b = idx >> 11;
    int j = idx & (GDIM - 1);
    int seg = j >> 9;
    int h = j & (HDIM - 1);
    int srcRow = (seg == 0) ? b : (seg == 1) ? (NCOMB + b)
               : (seg == 2) ? (BATCH + b)    : (NCOMB + BATCH + b);
    feat[idx] = (float)hS[(size_t)srcRow * HDIM + h];
}

// ---------------------------------------------------------------------------
// BatchNorm1d (training mode, biased variance) per column over 256 rows.
// ---------------------------------------------------------------------------
__global__ void bn_kernel(const float* __restrict__ X, int rows, int cols,
                          const float* __restrict__ gamma, const float* __restrict__ beta,
                          _Float16* __restrict__ outH, float* __restrict__ outF) {
    int col = blockIdx.x * blockDim.x + threadIdx.x;
    if (col >= cols) return;
    float s = 0.f, s2 = 0.f;
    for (int r = 0; r < rows; ++r) {
        float v = X[(size_t)r * cols + col];
        s += v; s2 += v * v;
    }
    float inv = 1.f / rows;
    float mu = s * inv;
    float var = s2 * inv - mu * mu;
    float sc = rsqrtf(var + EPS) * gamma[col];
    float sh = beta[col] - mu * sc;
    for (int r = 0; r < rows; ++r) {
        float v = X[(size_t)r * cols + col] * sc + sh;
        if (outH) outH[(size_t)r * cols + col] = (_Float16)v;
        else      outF[(size_t)r * cols + col] = v;
    }
}

// ---------------------------------------------------------------------------
// Final [256,2048] @ [3,2048]^T + bo -> [256,3]
// ---------------------------------------------------------------------------
__global__ void final_proj(const float* __restrict__ Y, const float* __restrict__ Wo,
                           const float* __restrict__ bo, float* __restrict__ out) {
    int idx = blockIdx.x * blockDim.x + threadIdx.x;
    if (idx >= BATCH * DOUT) return;
    int b = idx / DOUT;
    int d = idx % DOUT;
    const float* y = Y + (size_t)b * GDIM;
    const float* w = Wo + (size_t)d * GDIM;
    float acc = bo[d];
    for (int k = 0; k < GDIM; ++k) acc += y[k] * w[k];
    out[idx] = acc;
}

// ---------------------------------------------------------------------------
static inline void* carve(uintptr_t& p, size_t bytes) {
    void* r = (void*)p;
    p += (bytes + 255) & ~(size_t)255;
    return r;
}

extern "C" void kernel_launch(void* const* d_in, const int* in_sizes, int n_in,
                              void* d_out, int out_size, void* d_ws, size_t ws_size,
                              hipStream_t stream) {
    (void)in_sizes; (void)n_in; (void)out_size; (void)ws_size;

    const int*   prem  = (const int*)  d_in[0];
    const int*   hyp   = (const int*)  d_in[1];
    const float* embed = (const float*)d_in[2];
    const float* W_ih  = (const float*)d_in[3];
    const float* W_hh  = (const float*)d_in[4];
    const float* b_ih  = (const float*)d_in[5];
    const float* b_hh  = (const float*)d_in[6];
    const float* bn0_g = (const float*)d_in[7];
    const float* bn0_b = (const float*)d_in[8];
    const float* W0    = (const float*)d_in[9];
    const float* b0    = (const float*)d_in[10];
    const float* bn1_g = (const float*)d_in[11];
    const float* bn1_b = (const float*)d_in[12];
    const float* W1    = (const float*)d_in[13];
    const float* b1    = (const float*)d_in[14];
    const float* bno_g = (const float*)d_in[15];
    const float* bno_b = (const float*)d_in[16];
    const float* Wo    = (const float*)d_in[17];
    const float* bo    = (const float*)d_in[18];
    float* out = (float*)d_out;

    const size_t ROWS_X = (size_t)TSTEPS * NCOMB;   // 32768

    uintptr_t p = (uintptr_t)d_ws;
    _Float16* Xf16  = (_Float16*)carve(p, ROWS_X * EPAD * sizeof(_Float16));
    _Float16* WihH  = (_Float16*)carve(p, (size_t)GDIM * EPAD * sizeof(_Float16));
    _Float16* WhhH  = (_Float16*)carve(p, (size_t)GDIM * HDIM * sizeof(_Float16));
    _Float16* W0H   = (_Float16*)carve(p, (size_t)GDIM * GDIM * sizeof(_Float16));
    _Float16* W1H   = (_Float16*)carve(p, (size_t)GDIM * GDIM * sizeof(_Float16));
    _Float16* xproj = (_Float16*)carve(p, ROWS_X * GDIM * sizeof(_Float16));
    float*    Gbuf  = (float*)   carve(p, (size_t)NSTATE * GDIM * sizeof(float));
    _Float16* hS    = (_Float16*)carve(p, (size_t)NSTATE * HDIM * sizeof(_Float16));
    float*    cS    = (float*)   carve(p, (size_t)NSTATE * HDIM * sizeof(float));
    float*    feat  = (float*)   carve(p, (size_t)BATCH * GDIM * sizeof(float));
    _Float16* YH    = (_Float16*)carve(p, (size_t)BATCH * GDIM * sizeof(_Float16));
    float*    x1    = (float*)   carve(p, (size_t)BATCH * GDIM * sizeof(float));
    float*    x2    = (float*)   carve(p, (size_t)BATCH * GDIM * sizeof(float));
    float*    Ybn   = (float*)   carve(p, (size_t)BATCH * GDIM * sizeof(float));

    const int TB = 256;
    auto nb = [](size_t total, int tb) { return (unsigned)((total + tb - 1) / tb); };

    // 1) weight conversions to f16
    cvt_f16_pad<<<nb((size_t)GDIM * EPAD, TB), TB, 0, stream>>>(W_ih, WihH, GDIM, EDIM, EPAD);
    cvt_f16_pad<<<nb((size_t)GDIM * HDIM, TB), TB, 0, stream>>>(W_hh, WhhH, GDIM, HDIM, HDIM);
    cvt_f16_pad<<<nb((size_t)GDIM * GDIM, TB), TB, 0, stream>>>(W0, W0H, GDIM, GDIM, GDIM);
    cvt_f16_pad<<<nb((size_t)GDIM * GDIM, TB), TB, 0, stream>>>(W1, W1H, GDIM, GDIM, GDIM);

    // 2) embedding gather -> X f16
    gather_embed<<<nb(ROWS_X * EPAD, TB), TB, 0, stream>>>(prem, hyp, embed, Xf16);

    // 3) x_proj = X @ W_ih^T
    wmma_gemm<0><<<dim3((unsigned)(ROWS_X / BM), GDIM / BN), TB, 0, stream>>>(
        Xf16, EPAD, WihH, EPAD, xproj, GDIM, EPAD, nullptr, nullptr, nullptr, 0);

    // 4) init LSTM state
    (void)hipMemsetAsync(hS, 0, (size_t)NSTATE * HDIM * sizeof(_Float16), stream);
    (void)hipMemsetAsync(cS, 0, (size_t)NSTATE * HDIM * sizeof(float), stream);

    // 5) recurrence: fw rows 0..511 scan t, bw rows 512..1023 scan T-1-t
    for (int t = 0; t < TSTEPS; ++t) {
        wmma_gemm<1><<<dim3(NSTATE / BM, GDIM / BN), TB, 0, stream>>>(
            hS, HDIM, WhhH, HDIM, Gbuf, GDIM, HDIM, b_ih, b_hh, xproj, t);
        lstm_pointwise<<<nb((size_t)NSTATE * HDIM, TB), TB, 0, stream>>>(Gbuf, cS, hS);
    }

    // 6) feature concat
    build_feat<<<nb((size_t)BATCH * GDIM, TB), TB, 0, stream>>>(hS, feat);

    // 7) MLP
    bn_kernel<<<nb(GDIM, TB), TB, 0, stream>>>(feat, BATCH, GDIM, bn0_g, bn0_b, YH, nullptr);
    wmma_gemm<2><<<dim3(BATCH / BM, GDIM / BN), TB, 0, stream>>>(
        YH, GDIM, W0H, GDIM, x1, GDIM, GDIM, b0, nullptr, nullptr, 0);

    bn_kernel<<<nb(GDIM, TB), TB, 0, stream>>>(x1, BATCH, GDIM, bn1_g, bn1_b, YH, nullptr);
    wmma_gemm<2><<<dim3(BATCH / BM, GDIM / BN), TB, 0, stream>>>(
        YH, GDIM, W1H, GDIM, x2, GDIM, GDIM, b1, nullptr, nullptr, 0);

    bn_kernel<<<nb(GDIM, TB), TB, 0, stream>>>(x2, BATCH, GDIM, bno_g, bno_b, nullptr, Ybn);
    final_proj<<<nb((size_t)BATCH * DOUT, TB), TB, 0, stream>>>(Ybn, Wo, bo, out);
}
